// Proposal1Model_85237920956697
// MI455X (gfx1250) — matrix operand order repaired and checked by reading, not compiled, not run on hardware
//
#include <hip/hip_runtime.h>
#include <hip/hip_bf16.h>
#include <math.h>

// ---------------------------------------------------------------------------
// Problem constants (from reference)
// ---------------------------------------------------------------------------
#define BATCH   8192
#define SEQL    64
#define HID     64          // H
#define G3      192         // 3*H
#define EDIM    32
#define SCTX    4096        // S1 == S2
#define KSEL    20

typedef __attribute__((ext_vector_type(16))) _Float16 v16h;
typedef __attribute__((ext_vector_type(8)))  _Float16 v8h;
typedef __attribute__((ext_vector_type(8)))  float    v8f;
typedef __attribute__((ext_vector_type(4)))  float    v4f;

// IR-level optimization fence: blocks LICM/cross-iteration code motion while
// leaving every LDS access a plain typed load (full ds_load_b128 codegen).
#define OPT_FENCE() asm volatile("" ::: "memory")

__device__ __forceinline__ float fast_rcp(float x) {
#if __has_builtin(__builtin_amdgcn_rcpf)
    return __builtin_amdgcn_rcpf(x);
#else
    return 1.0f / x;
#endif
}

__device__ __forceinline__ float sigmoidf_(float x) {
    return fast_rcp(1.0f + __expf(-x));
}

__device__ __forceinline__ float tanhf_(float x) {
    x = fminf(fmaxf(x, -15.0f), 15.0f);
    float e = __expf(2.0f * x);
    return (e - 1.0f) * fast_rcp(e + 1.0f);
}

__device__ __forceinline__ v8f wmma_f16(v16h a, v16h b, v8f c) {
    return __builtin_amdgcn_wmma_f32_16x16x32_f16(
        /*neg_a=*/false, a, /*neg_b=*/false, b,
        /*c_mod=*/(short)0, c, /*reuse_a=*/false, /*reuse_b=*/false);
}

// first WMMA of an accumulation chain: C = 0 (folds to inline constant 0)
__device__ __forceinline__ v8f wmma_f16_z(v16h a, v16h b) {
    v8f z = {};
    return __builtin_amdgcn_wmma_f32_16x16x32_f16(
        false, a, false, b, (short)0, z, false, false);
}

// A-fragment (16x32 f16) gather from a row-major 16x64 f16 LDS tile.
// ISA layout: lanes 0-15 (row m): K = base+{0..7}, base+16+{0..7}
//             lanes 16-31 (row m): K = base+8+{0..7}, base+24+{0..7}
__device__ __forceinline__ v16h load_afrag(const _Float16* hb, int m, int hi, int kc) {
    const int base = kc * 32 + hi * 8;
    v8h a0 = *(const v8h*)(hb + m * HID + base);
    v8h a1 = *(const v8h*)(hb + m * HID + base + 16);
    return __builtin_shufflevector(a0, a1, 0,1,2,3,4,5,6,7,8,9,10,11,12,13,14,15);
}

// B-fragment (32x16 f16) — weights pre-swizzled fragment-major in LDS:
// fragment (nt,kc): 32 lanes x 16 contiguous halves (two ds_load_b128).
__device__ __forceinline__ v16h load_bfrag(const _Float16* fb, int nt, int kc, int lane) {
    const v8h* p = (const v8h*)(fb + (((nt << 1) | kc) * 32 + lane) * 16);
    v8h b0 = p[0];
    v8h b1 = p[1];
    return __builtin_shufflevector(b0, b1, 0,1,2,3,4,5,6,7,8,9,10,11,12,13,14,15);
}

// ---------------------------------------------------------------------------
// Kernel 1: fused 2-layer GRU over all 64 steps.  grid=(B/128, 2 sides),
// block=256 (8 waves, 16 batch rows per wave).  WMMA f32_16x16x32_f16.
// Dynamic LDS layout (halves unless noted):
//   [0)        Whh0 fragments   12288
//   [12288)    Wih1 fragments   12288
//   [24576)    Whh1 fragments   12288
//   [36864)    h0 tiles         8*1024
//   [45056)    h1 tiles         8*1024
//   [53248)    f32 region: biases 5*192, x tile (time-major) 64*128
// ---------------------------------------------------------------------------
extern "C" __global__ __launch_bounds__(256, 1)
void gru2_kernel(const float* __restrict__ xL, const float* __restrict__ xR,
                 const float* glWih0, const float* glWhh0, const float* glbih0, const float* glbhh0,
                 const float* glWih1, const float* glWhh1, const float* glbih1, const float* glbhh1,
                 const float* grWih0, const float* grWhh0, const float* grbih0, const float* grbhh0,
                 const float* grWih1, const float* grWhh1, const float* grbih1, const float* grbhh1,
                 float* __restrict__ outLR)
{
    extern __shared__ _Float16 smem[];
    const int tid  = threadIdx.x;
    const int wave = tid >> 5;
    const int lane = tid & 31;
    const int nc   = lane & 15;         // N column (or row M for A frags)
    const int hi   = lane >> 4;

    const int side = blockIdx.y;
    const float* x    = side ? xR     : xL;
    const float* Wih0 = side ? grWih0 : glWih0;
    const float* Whh0 = side ? grWhh0 : glWhh0;
    const float* bih0 = side ? grbih0 : glbih0;
    const float* bhh0 = side ? grbhh0 : glbhh0;
    const float* Wih1 = side ? grWih1 : glWih1;
    const float* Whh1 = side ? grWhh1 : glWhh1;
    const float* bih1 = side ? grbih1 : glbih1;
    const float* bhh1 = side ? grbhh1 : glbhh1;

    _Float16* whh0f = smem;                 // 12288 halves
    _Float16* wih1f = smem + 12288;
    _Float16* whh1f = smem + 24576;
    _Float16* h0b   = smem + 36864 + wave * 1024;   // 16x64 per wave
    _Float16* h1b   = smem + 36864 + 8192 + wave * 1024;
    float*    fb    = (float*)(smem + 53248);        // biases then x tile
    float*    xls   = fb + 5 * G3;                   // 64 steps x 128 rows

    // ---- stage weights (as WMMA B-fragments, f32 -> f16) ----
    for (int i = tid; i < 3 * 12288; i += 256) {
        int m = i / 12288;
        int r = i % 12288;
        int e = r & 15;
        int l = (r >> 4) & 31;
        int f = r >> 9;                 // 0..23 = nt*2+kc
        int nt = f >> 1, kc = f & 1;
        int n = nt * 16 + (l & 15);
        int k = kc * 32 + (l >> 4) * 8 + (e & 7) + (e >> 3) * 16;
        const float* W = (m == 0) ? Whh0 : (m == 1) ? Wih1 : Whh1;
        smem[i] = (_Float16)W[n * HID + k];
    }
    // ---- stage biases + wih0 column (f32) ----
    if (tid < G3) {
        fb[          tid] = bhh0[tid];
        fb[    G3 +  tid] = bih0[tid];
        fb[2 * G3 +  tid] = Wih0[tid];
        fb[3 * G3 +  tid] = bih1[tid];
        fb[4 * G3 +  tid] = bhh1[tid];
    }
    // ---- stage x tile transposed to time-major (coalesced global reads) ----
    {
        const size_t gbase = (size_t)blockIdx.x * 128 * SEQL;
        for (int i = tid; i < 128 * SEQL; i += 256) {
            int row = i >> 6, tt = i & 63;
            xls[tt * 128 + row] = x[gbase + i];
        }
    }
    // ---- zero hidden-state tiles (per wave region) ----
    for (int i = lane; i < 1024; i += 32) { h0b[i] = (_Float16)0.0f; h1b[i] = (_Float16)0.0f; }
    __syncthreads();

    const int rowbase = (blockIdx.x * 8 + wave) * 16;
    const int xoff = wave * 16 + 8 * hi;          // this lane's 8 consecutive rows

    float h0r[4][8], h1r[4][8];
#pragma unroll
    for (int j = 0; j < 4; ++j)
#pragma unroll
        for (int r = 0; r < 8; ++r) { h0r[j][r] = 0.0f; h1r[j][r] = 0.0f; }

#pragma unroll 1
    for (int t = 0; t < SEQL; ++t) {
        OPT_FENCE();   // keep per-iteration LDS loads in the loop (no LICM hoist)

        // 8 consecutive rows -> two ds_load_b128
        float xv[8];
        {
            const v4f* xp4 = (const v4f*)(xls + t * 128 + xoff);
            v4f x0 = xp4[0], x1 = xp4[1];
#pragma unroll
            for (int r = 0; r < 4; ++r) { xv[r] = x0[r]; xv[r + 4] = x1[r]; }
        }

        // ===== layer 0:  gh = h0 @ Whh0^T ;  gi = xt*wih0 + bih0 =====
        v16h a00 = load_afrag(h0b, nc, hi, 0);
        v16h a01 = load_afrag(h0b, nc, hi, 1);
#pragma unroll
        for (int j = 0; j < 4; ++j) {
            const float bhr  = fb[(j    ) * 16 + nc];
            const float bhz  = fb[(j + 4) * 16 + nc];
            const float bhn  = fb[(j + 8) * 16 + nc];
            const float bir  = fb[G3 + (j    ) * 16 + nc];
            const float biz  = fb[G3 + (j + 4) * 16 + nc];
            const float bin_ = fb[G3 + (j + 8) * 16 + nc];
            const float wir  = fb[2 * G3 + (j    ) * 16 + nc];
            const float wiz  = fb[2 * G3 + (j + 4) * 16 + nc];
            const float win  = fb[2 * G3 + (j + 8) * 16 + nc];
            const float bsr = bir + bhr;          // r-gate bias sum
            const float bsz = biz + bhz;          // z-gate bias sum
            v8f ghr = wmma_f16_z(a00, load_bfrag(whh0f, j,     0, lane));
            ghr     = wmma_f16 (a01, load_bfrag(whh0f, j,     1, lane), ghr);
            v8f ghz = wmma_f16_z(a00, load_bfrag(whh0f, j + 4, 0, lane));
            ghz     = wmma_f16 (a01, load_bfrag(whh0f, j + 4, 1, lane), ghz);
            v8f ghn = wmma_f16_z(a00, load_bfrag(whh0f, j + 8, 0, lane));
            ghn     = wmma_f16 (a01, load_bfrag(whh0f, j + 8, 1, lane), ghn);
#pragma unroll
            for (int r = 0; r < 8; ++r) {
                float rg = sigmoidf_(xv[r] * wir + bsr + ghr[r]);
                float zg = sigmoidf_(xv[r] * wiz + bsz + ghz[r]);
                float ng = tanhf_(xv[r] * win + bin_ + rg * (ghn[r] + bhn));
                h0r[j][r] = (1.0f - zg) * ng + zg * h0r[j][r];
                h0b[(r + 8 * hi) * HID + j * 16 + nc] = (_Float16)h0r[j][r];
            }
        }

        OPT_FENCE();   // bound scheduling window between the two layers

        // ===== layer 1:  gi = y0_t @ Wih1^T ;  gh = h1 @ Whh1^T =====
        v16h ax0 = load_afrag(h0b, nc, hi, 0);   // fresh h0 (same wave; ds-ordered)
        v16h ax1 = load_afrag(h0b, nc, hi, 1);
        v16h ah0 = load_afrag(h1b, nc, hi, 0);
        v16h ah1 = load_afrag(h1b, nc, hi, 1);
#pragma unroll
        for (int j = 0; j < 4; ++j) {
            const float i_r = fb[3 * G3 + (j    ) * 16 + nc];
            const float i_z = fb[3 * G3 + (j + 4) * 16 + nc];
            const float i_n = fb[3 * G3 + (j + 8) * 16 + nc];
            const float h_r = fb[4 * G3 + (j    ) * 16 + nc];
            const float h_z = fb[4 * G3 + (j + 4) * 16 + nc];
            const float h_n = fb[4 * G3 + (j + 8) * 16 + nc];
            const float bsr = i_r + h_r;
            const float bsz = i_z + h_z;
            v8f gir_ = wmma_f16_z(ax0, load_bfrag(wih1f, j,     0, lane));
            gir_     = wmma_f16 (ax1, load_bfrag(wih1f, j,     1, lane), gir_);
            v8f giz_ = wmma_f16_z(ax0, load_bfrag(wih1f, j + 4, 0, lane));
            giz_     = wmma_f16 (ax1, load_bfrag(wih1f, j + 4, 1, lane), giz_);
            v8f gin_ = wmma_f16_z(ax0, load_bfrag(wih1f, j + 8, 0, lane));
            gin_     = wmma_f16 (ax1, load_bfrag(wih1f, j + 8, 1, lane), gin_);
            v8f ghr_ = wmma_f16_z(ah0, load_bfrag(whh1f, j,     0, lane));
            ghr_     = wmma_f16 (ah1, load_bfrag(whh1f, j,     1, lane), ghr_);
            v8f ghz_ = wmma_f16_z(ah0, load_bfrag(whh1f, j + 4, 0, lane));
            ghz_     = wmma_f16 (ah1, load_bfrag(whh1f, j + 4, 1, lane), ghz_);
            v8f ghn_ = wmma_f16_z(ah0, load_bfrag(whh1f, j + 8, 0, lane));
            ghn_     = wmma_f16 (ah1, load_bfrag(whh1f, j + 8, 1, lane), ghn_);
#pragma unroll
            for (int r = 0; r < 8; ++r) {
                float rg = sigmoidf_(gir_[r] + ghr_[r] + bsr);
                float zg = sigmoidf_(giz_[r] + ghz_[r] + bsz);
                float ng = tanhf_((gin_[r] + i_n) + rg * (ghn_[r] + h_n));
                h1r[j][r] = (1.0f - zg) * ng + zg * h1r[j][r];
                h1b[(r + 8 * hi) * HID + j * 16 + nc] = (_Float16)h1r[j][r];
            }
        }
    }

    // write final h1 (C layout -> row-major global)
#pragma unroll
    for (int j = 0; j < 4; ++j)
#pragma unroll
        for (int r = 0; r < 8; ++r) {
            int row = rowbase + r + 8 * hi;
            outLR[((size_t)side * BATCH + row) * HID + j * 16 + nc] = h1r[j][r];
        }
}

// ---------------------------------------------------------------------------
// Kernel 2: KNN features per row.  grid=(B, 2 contexts), block=64.
// Distances in LDS, 21 deterministic argmax-extraction passes.
// ---------------------------------------------------------------------------
extern "C" __global__ void knn_kernel(const int* __restrict__ idx1, const int* __restrict__ idx2,
                                      const float* __restrict__ yc1, const float* __restrict__ yc2,
                                      const float* __restrict__ emb1, const float* __restrict__ emb2,
                                      float* __restrict__ feats)
{
    __shared__ float dist[SCTX];
    __shared__ float evec[EDIM];
    __shared__ float wbuf[KSEL];
    __shared__ float selbuf[KSEL];
    __shared__ float redv[64];
    __shared__ int   redi[64];

    const int tid = threadIdx.x;
    const int b   = blockIdx.x;
    const int c   = blockIdx.y;
    const int*   idx = c ? idx2 : idx1;
    const float* emb = c ? emb2 : emb1;
    const float* yc  = c ? yc2  : yc1;

    const int e0 = idx[b];
    if (tid < EDIM) evec[tid] = emb[(size_t)e0 * EDIM + tid];
    __syncthreads();

    float en2 = 0.0f;
#pragma unroll
    for (int k = 0; k < EDIM; ++k) en2 += evec[k] * evec[k];

    for (int j = tid; j < SCTX; j += 64) {
        float dot = 0.0f, n2 = 0.0f;
        const float* ej = emb + (size_t)j * EDIM;
#pragma unroll
        for (int k = 0; k < EDIM; ++k) { float v = ej[k]; dot += v * evec[k]; n2 += v * v; }
        float d2 = en2 + n2 - 2.0f * dot;
        dist[j] = sqrtf(fmaxf(d2, 1e-12f));
    }
    __syncthreads();

    // w uses FIXED columns S-21..S-2 of sim (reference quirk)
    if (tid < KSEL) wbuf[tid] = __expf(-dist[SCTX - (KSEL + 1) + tid]);   // TAU = 1
    __syncthreads();

    for (int p = 0; p < KSEL + 1; ++p) {
        float bv = -1.0f; int bi = -1;
        for (int j = tid; j < SCTX; j += 64) {
            float v = dist[j];
            if (v > bv || (v == bv && j > bi)) { bv = v; bi = j; }
        }
        redv[tid] = bv; redi[tid] = bi;
        __syncthreads();
        if (tid == 0) {
            float mv = -2.0f; int mi = -1;
            for (int q = 0; q < 64; ++q)
                if (redv[q] > mv) { mv = redv[q]; mi = redi[q]; }
            dist[mi] = -1.0f;
            if (p >= 1) selbuf[KSEL - p] = yc[(size_t)b * SCTX + mi];  // rank S-1-p -> k=20-p
        }
        __syncthreads();
    }

    if (tid == 0) {
        float s = 0.0f, wsum = 0.0f, wm = 0.0f;
        for (int k = 0; k < KSEL; ++k) { s += selbuf[k]; wsum += wbuf[k]; wm += selbuf[k] * wbuf[k]; }
        float mean = s / (float)KSEL;
        float ssq = 0.0f;
        for (int k = 0; k < KSEL; ++k) { float d = selbuf[k] - mean; ssq += d * d; }
        float stdv = sqrtf(ssq / (float)(KSEL - 1));
        float* f = feats + ((size_t)c * BATCH + b) * 3;
        f[0] = wm / wsum; f[1] = wsum; f[2] = stdv;
    }
}

// ---------------------------------------------------------------------------
// Kernel 3: per-row heads + per-block deterministic loss partials.
// grid=64, block=128 (one row per thread).
// ---------------------------------------------------------------------------
extern "C" __global__ void head_kernel(const float* __restrict__ y, const float* __restrict__ outLR,
                                       const float* __restrict__ feats,
                                       const float* mean_W, const float* mean_b,
                                       const float* std_W,  const float* std_b,
                                       const float* ol1_W,  const float* ol1_b,
                                       const float* om_W,   const float* om_b,
                                       const float* os_W,   const float* os_b,
                                       float* __restrict__ partials)
{
    __shared__ float r1[128], r2[128];
    const int tid = threadIdx.x;
    const int row = blockIdx.x * 128 + tid;

    const float* ol  = outLR + (size_t)row * HID;
    const float* orr = outLR + ((size_t)BATCH + row) * HID;
    float mts = mean_b[0], sts = std_b[0];
#pragma unroll 8
    for (int j = 0; j < HID; ++j) {
        mts += ol[j] * mean_W[j] + orr[j] * mean_W[HID + j];
        sts += ol[j] * std_W[j]  + orr[j] * std_W[HID + j];
    }

    float f[8];
    f[0] = feats[(size_t)row * 3 + 0];
    f[1] = feats[(size_t)row * 3 + 1];
    f[2] = feats[(size_t)row * 3 + 2];
    f[3] = feats[((size_t)BATCH + row) * 3 + 0];
    f[4] = feats[((size_t)BATCH + row) * 3 + 1];
    f[5] = feats[((size_t)BATCH + row) * 3 + 2];
    f[6] = mts; f[7] = sts;

    float mo = om_b[0], so = os_b[0];
    for (int i = 0; i < 64; ++i) {
        float hv = ol1_b[i];
#pragma unroll
        for (int j = 0; j < 8; ++j) hv += ol1_W[i * 8 + j] * f[j];
        hv = fmaxf(hv, 0.0f);
        mo += om_W[i] * hv;
        so += os_W[i] * hv;
    }

    float yb = y[row];
    float d1 = yb - mts, d2 = yb - mo;
    r1[tid] = d1 * d1 / __expf(sts) + sts;
    r2[tid] = d2 * d2 / __expf(so) + so;
    __syncthreads();
    for (int s = 64; s > 0; s >>= 1) {
        if (tid < s) { r1[tid] += r1[tid + s]; r2[tid] += r2[tid + s]; }
        __syncthreads();
    }
    if (tid == 0) { partials[blockIdx.x * 2 + 0] = r1[0]; partials[blockIdx.x * 2 + 1] = r2[0]; }
}

extern "C" __global__ void finish_kernel(const float* __restrict__ partials, float* __restrict__ out)
{
    if (threadIdx.x == 0) {
        float s1 = 0.0f, s2 = 0.0f;
        for (int i = 0; i < 64; ++i) { s1 += partials[2 * i]; s2 += partials[2 * i + 1]; }
        out[0] = s1 / (float)BATCH;
        out[1] = s2 / (float)BATCH;
    }
}

// ---------------------------------------------------------------------------
// Host-side launcher
// ---------------------------------------------------------------------------
extern "C" void kernel_launch(void* const* d_in, const int* in_sizes, int n_in,
                              void* d_out, int out_size, void* d_ws, size_t ws_size,
                              hipStream_t stream)
{
    const float* x_left  = (const float*)d_in[0];
    const float* x_right = (const float*)d_in[1];
    const float* y       = (const float*)d_in[2];
    const int*   index1  = (const int*)  d_in[3];
    const int*   index2  = (const int*)  d_in[4];
    const float* y1c     = (const float*)d_in[5];
    const float* y2c     = (const float*)d_in[6];
    const float* emb1    = (const float*)d_in[7];
    const float* emb2    = (const float*)d_in[8];
    const float* glWih0  = (const float*)d_in[9];
    const float* glWhh0  = (const float*)d_in[10];
    const float* glbih0  = (const float*)d_in[11];
    const float* glbhh0  = (const float*)d_in[12];
    const float* glWih1  = (const float*)d_in[13];
    const float* glWhh1  = (const float*)d_in[14];
    const float* glbih1  = (const float*)d_in[15];
    const float* glbhh1  = (const float*)d_in[16];
    const float* grWih0  = (const float*)d_in[17];
    const float* grWhh0  = (const float*)d_in[18];
    const float* grbih0  = (const float*)d_in[19];
    const float* grbhh0  = (const float*)d_in[20];
    const float* grWih1  = (const float*)d_in[21];
    const float* grWhh1  = (const float*)d_in[22];
    const float* grbih1  = (const float*)d_in[23];
    const float* grbhh1  = (const float*)d_in[24];
    const float* mean_W  = (const float*)d_in[25];
    const float* mean_b  = (const float*)d_in[26];
    const float* std_W   = (const float*)d_in[27];
    const float* std_b   = (const float*)d_in[28];
    const float* ol1_W   = (const float*)d_in[29];
    const float* ol1_b   = (const float*)d_in[30];
    const float* om_W    = (const float*)d_in[31];
    const float* om_b    = (const float*)d_in[32];
    const float* os_W    = (const float*)d_in[33];
    const float* os_b    = (const float*)d_in[34];

    float* outLR    = (float*)d_ws;                       // 2*B*H
    float* feats    = outLR + 2 * (size_t)BATCH * HID;    // 2*B*3
    float* partials = feats + 2 * (size_t)BATCH * 3;      // 128

    // GRU dynamic LDS: 53248 halves + (5*192 + 64*128) floats = 143104 bytes
    const size_t gru_lds = (size_t)53248 * sizeof(_Float16)
                         + (size_t)(5 * G3 + SEQL * 128) * sizeof(float);
    gru2_kernel<<<dim3(BATCH / 128, 2), 256, gru_lds, stream>>>(
        x_left, x_right,
        glWih0, glWhh0, glbih0, glbhh0, glWih1, glWhh1, glbih1, glbhh1,
        grWih0, grWhh0, grbih0, grbhh0, grWih1, grWhh1, grbih1, grbhh1,
        outLR);

    knn_kernel<<<dim3(BATCH, 2), 64, 0, stream>>>(index1, index2, y1c, y2c, emb1, emb2, feats);

    head_kernel<<<dim3(64), 128, 0, stream>>>(y, outLR, feats,
        mean_W, mean_b, std_W, std_b, ol1_W, ol1_b, om_W, om_b, os_W, os_b, partials);

    finish_kernel<<<dim3(1), 32, 0, stream>>>(partials, (float*)d_out);
}
